// SelfAttention_11381663334409
// MI455X (gfx1250) — matrix operand order repaired and checked by reading, not compiled, not run on hardware
//
#include <hip/hip_runtime.h>

#define BATCH 4
#define SEQ   2048
#define HID   1024

typedef __bf16  bf16_t;
typedef bf16_t  v16bf __attribute__((ext_vector_type(16)));
typedef float   v8f   __attribute__((ext_vector_type(8)));
typedef unsigned u32x4 __attribute__((ext_vector_type(4)));
typedef unsigned u32x8 __attribute__((ext_vector_type(8)));

union FragAB { v16bf v; uint4 q[2]; };

// ---------- fp32 <-> bf16 helpers (RNE) ----------
__device__ __forceinline__ unsigned short f2bf1(float f) {
  union { float f; unsigned u; } x; x.f = f;
  unsigned r = x.u + 0x7FFFu + ((x.u >> 16) & 1u);
  return (unsigned short)(r >> 16);
}
__device__ __forceinline__ unsigned f2bf_pk(float lo, float hi) {
  union { float f; unsigned u; } a, b; a.f = lo; b.f = hi;
  unsigned ra = a.u + 0x7FFFu + ((a.u >> 16) & 1u);
  unsigned rb = b.u + 0x7FFFu + ((b.u >> 16) & 1u);
  return (ra >> 16) | (rb & 0xFFFF0000u);
}
__device__ __forceinline__ float bf2f(unsigned short s) {
  union { unsigned u; float f; } x; x.u = ((unsigned)s) << 16; return x.f;
}

enum { OUT_F32 = 0, OUT_BF16 = 1, OUT_BF16_T = 2 };

// ---------- fragment math shared by all GEMM cores ----------
__device__ __forceinline__ void compute_step(
    const unsigned short* __restrict__ as, const unsigned short* __restrict__ bs,
    int waveM, int waveN, int r, int hi, v8f acc[4][2])
{
  FragAB a[4], b[2];
#pragma unroll
  for (int mi = 0; mi < 4; ++mi) {
    int row = waveM * 64 + mi * 16 + r;
    a[mi].q[0] = *(const uint4*)(as + row * 40 + hi * 8);
    a[mi].q[1] = *(const uint4*)(as + row * 40 + 16 + hi * 8);
  }
#pragma unroll
  for (int ni = 0; ni < 2; ++ni) {
    int n = waveN * 32 + ni * 16 + r;
    b[ni].q[0] = *(const uint4*)(bs + n * 40 + hi * 8);
    b[ni].q[1] = *(const uint4*)(bs + n * 40 + 16 + hi * 8);
  }
#pragma unroll
  for (int mi = 0; mi < 4; ++mi)
#pragma unroll
    for (int ni = 0; ni < 2; ++ni)
      acc[mi][ni] = __builtin_amdgcn_wmma_f32_16x16x32_bf16(
          false, a[mi].v, false, b[ni].v, (short)0, acc[mi][ni], false, false);
}

template<int OMODE>
__device__ __forceinline__ void epilogue(
    v8f acc[4][2], void* __restrict__ outp, int ldc, float scale,
    int n0, int m0out, int waveM, int waveN, int r, int hi)
{
#pragma unroll
  for (int mi = 0; mi < 4; ++mi) {
#pragma unroll
    for (int ni = 0; ni < 2; ++ni) {
      int col     = n0 + waveN * 32 + ni * 16 + r;
      int rowbase = waveM * 64 + mi * 16 + hi * 8;
#pragma unroll
      for (int rr = 0; rr < 8; ++rr) {
        int   row = rowbase + rr;
        float val = acc[mi][ni][rr] * scale;
        if (OMODE == OUT_F32) {
          ((float*)outp)[(size_t)(m0out + row) * ldc + col] = val;
        } else if (OMODE == OUT_BF16) {
          ((unsigned short*)outp)[(size_t)(m0out + row) * ldc + col] = f2bf1(val);
        } else { // transposed bf16 store
          ((unsigned short*)outp)[(size_t)col * ldc + m0out + row] = f2bf1(val);
        }
      }
    }
  }
}

// ========== core 1: fp32-source GEMM (convert in regs, sync LDS staging) ==========
__device__ __forceinline__ void stage128x32_f32(const float* __restrict__ src, int ld,
                                                int row0, int k0,
                                                unsigned short* dst, int tid) {
#pragma unroll
  for (int i = 0; i < 4; ++i) {
    int e   = tid + i * 256;
    int row = e >> 3;
    int col = (e & 7) << 2;
    const float4 f = *(const float4*)(src + (size_t)(row0 + row) * ld + k0 + col);
    uint2 st; st.x = f2bf_pk(f.x, f.y); st.y = f2bf_pk(f.z, f.w);
    *(uint2*)(dst + row * 40 + col) = st;
  }
}

template<int OMODE>
__device__ __forceinline__ void gemm_f32src(
    const float* __restrict__ A, int lda,
    const float* __restrict__ B, int ldb,
    int K, void* __restrict__ outp, int ldc,
    float scale, int m0, int n0, int m0out)
{
  __shared__ unsigned short As[128 * 40];
  __shared__ unsigned short Bs[128 * 40];

  const int tid = threadIdx.x, lane = tid & 31, wave = tid >> 5;
  const int waveM = wave >> 2, waveN = wave & 3, r = lane & 15, hi = lane >> 4;

  v8f acc[4][2];
#pragma unroll
  for (int mi = 0; mi < 4; ++mi)
#pragma unroll
    for (int ni = 0; ni < 2; ++ni)
#pragma unroll
      for (int t = 0; t < 8; ++t) acc[mi][ni][t] = 0.0f;

  for (int k0 = 0; k0 < K; k0 += 32) {
    stage128x32_f32(A, lda, m0, k0, As, tid);
    stage128x32_f32(B, ldb, n0, k0, Bs, tid);
    __syncthreads();
    compute_step(As, Bs, waveM, waveN, r, hi, acc);
    __syncthreads();
  }
  epilogue<OMODE>(acc, outp, ldc, scale, n0, m0out, waveM, waveN, r, hi);
}

// ========== core 2: bf16-source GEMM, double-buffered ASYNC memory->LDS ==========
template<int OMODE>
__device__ __forceinline__ void gemm_bf16_async(
    const unsigned short* __restrict__ A, int lda,
    const unsigned short* __restrict__ B, int ldb,
    int K, void* __restrict__ outp, int ldc,
    float scale, int m0, int n0)
{
  __shared__ unsigned short As[2][128 * 40];
  __shared__ unsigned short Bs[2][128 * 40];

  const int tid = threadIdx.x, lane = tid & 31, wave = tid >> 5;
  const int waveM = wave >> 2, waveN = wave & 3, r = lane & 15, hi = lane >> 4;

  v8f acc[4][2];
#pragma unroll
  for (int mi = 0; mi < 4; ++mi)
#pragma unroll
    for (int ni = 0; ni < 2; ++ni)
#pragma unroll
      for (int t = 0; t < 8; ++t) acc[mi][ni][t] = 0.0f;

  auto issue_tile = [&](int k0, int buf) {
#pragma unroll
    for (int i = 0; i < 2; ++i) {
      int e = tid + i * 256, row = e >> 2, col = (e & 3) << 3;
      const unsigned short* ga = A + (size_t)(m0 + row) * lda + k0 + col;
      unsigned la = (unsigned)(unsigned long long)(const void*)&As[buf][row * 40 + col];
      asm volatile("global_load_async_to_lds_b128 %0, %1, off"
                   :: "v"(la), "v"((unsigned long long)ga) : "memory");
      const unsigned short* gb = B + (size_t)(n0 + row) * ldb + k0 + col;
      unsigned lb = (unsigned)(unsigned long long)(const void*)&Bs[buf][row * 40 + col];
      asm volatile("global_load_async_to_lds_b128 %0, %1, off"
                   :: "v"(lb), "v"((unsigned long long)gb) : "memory");
    }
  };

  const int nk = K >> 5;
  issue_tile(0, 0);
  for (int i = 0; i < nk; ++i) {
    if (i + 1 < nk) {
      issue_tile((i + 1) << 5, (i + 1) & 1);
      asm volatile("s_wait_asynccnt 0x4" ::: "memory");
    } else {
      asm volatile("s_wait_asynccnt 0x0" ::: "memory");
    }
    __syncthreads();
    compute_step(As[i & 1], Bs[i & 1], waveM, waveN, r, hi, acc);
    __syncthreads();
  }
  epilogue<OMODE>(acc, outp, ldc, scale, n0, m0, waveM, waveN, r, hi);
}

// ========== core 3: bf16-source GEMM, Tensor Data Mover double-buffered ==========
// D# for a 128x32 bf16 tile into padded LDS rows (32 data + 8 pad shorts):
//   data_size=2B, tile_dim0=32, tile_dim1=128, dim0_stride=ld,
//   pad_enable, pad_interval=3 (16 DWORDs), pad_amount=3 (4 DWORDs)
__device__ __forceinline__ void tdm_load_tile_bf16(
    const unsigned short* gsrc, unsigned ld, unsigned lds_off)
{
  unsigned long long ga = (unsigned long long)gsrc;
  u32x4 g0;
  g0[0] = 1u;                                   // count=1 valid user descriptor
  g0[1] = lds_off;                              // lds_addr
  g0[2] = (unsigned)ga;                         // global_addr[31:0]
  g0[3] = (unsigned)(ga >> 32) | 0x80000000u;   // global_addr[56:32] | type=2
  u32x8 g1;
  g1[0] = (1u << 16) | (1u << 20) | (3u << 22) | (3u << 25);
  g1[1] = (ld & 0xFFFFu) << 16;                 // tensor_dim0[15:0]
  g1[2] = (ld >> 16);                           // tensor_dim0[31:16] | tensor_dim1[15:0]=0
  g1[3] = 1u | (32u << 16);                     // tensor_dim1[31:16]=1 (64K rows) | tile_dim0=32
  g1[4] = 128u;                                 // tile_dim1=128, tile_dim2=0
  g1[5] = ld;                                   // tensor_dim0_stride[31:0]
  g1[6] = 0u;                                   // stride[47:32]=0, dim1_stride lo=0
  g1[7] = 0u;
  asm volatile("tensor_load_to_lds %0, %1" :: "s"(g0), "s"(g1) : "memory");
}

template<int OMODE>
__device__ __forceinline__ void gemm_bf16_tdm(
    const unsigned short* __restrict__ A, int lda,
    const unsigned short* __restrict__ B, int ldb,
    int K, void* __restrict__ outp, int ldc,
    float scale, int m0, int n0)
{
  __shared__ unsigned short As[2][128 * 40];
  __shared__ unsigned short Bs[2][128 * 40];

  const int tid = threadIdx.x, lane = tid & 31, wave = tid >> 5;
  const int waveM = wave >> 2, waveN = wave & 3, r = lane & 15, hi = lane >> 4;

  v8f acc[4][2];
#pragma unroll
  for (int mi = 0; mi < 4; ++mi)
#pragma unroll
    for (int ni = 0; ni < 2; ++ni)
#pragma unroll
      for (int t = 0; t < 8; ++t) acc[mi][ni][t] = 0.0f;

  const unsigned as_off[2] = { (unsigned)(unsigned long long)(const void*)&As[0][0],
                               (unsigned)(unsigned long long)(const void*)&As[1][0] };
  const unsigned bs_off[2] = { (unsigned)(unsigned long long)(const void*)&Bs[0][0],
                               (unsigned)(unsigned long long)(const void*)&Bs[1][0] };

  const int nk = K >> 5;
  if (wave == 0) {
    tdm_load_tile_bf16(A + (size_t)m0 * lda, (unsigned)lda, as_off[0]);
    tdm_load_tile_bf16(B + (size_t)n0 * ldb, (unsigned)ldb, bs_off[0]);
  }
  for (int i = 0; i < nk; ++i) {
    if (wave == 0) {
      if (i + 1 < nk) {
        int k1 = (i + 1) << 5, nb = (i + 1) & 1;
        tdm_load_tile_bf16(A + (size_t)m0 * lda + k1, (unsigned)lda, as_off[nb]);
        tdm_load_tile_bf16(B + (size_t)n0 * ldb + k1, (unsigned)ldb, bs_off[nb]);
        __builtin_amdgcn_s_wait_tensorcnt(2);  // tile i landed; tile i+1 in flight
      } else {
        __builtin_amdgcn_s_wait_tensorcnt(0);
      }
    }
    __syncthreads();
    compute_step(As[i & 1], Bs[i & 1], waveM, waveN, r, hi, acc);
    __syncthreads();
  }
  epilogue<OMODE>(acc, outp, ldc, scale, n0, m0, waveM, waveN, r, hi);
}

// ---------- kernel 1: fused QKV projection (y = x @ W^T), v stored transposed ----------
__global__ __launch_bounds__(256) void qkv_kernel(
    const float* __restrict__ q, const float* __restrict__ k, const float* __restrict__ v,
    const float* __restrict__ WQ, const float* __restrict__ WK, const float* __restrict__ WV,
    unsigned short* __restrict__ qb, unsigned short* __restrict__ kb,
    unsigned short* __restrict__ vt)
{
  const int m0 = blockIdx.x * 128;
  const int n0 = blockIdx.y * 128;
  const int z  = blockIdx.z;
  if (z == 0) {
    gemm_f32src<OUT_BF16>(q, HID, WQ, HID, HID, qb, HID, 1.0f, m0, n0, m0);
  } else if (z == 1) {
    gemm_f32src<OUT_BF16>(k, HID, WK, HID, HID, kb, HID, 1.0f, m0, n0, m0);
  } else {
    unsigned short* outb = vt + (size_t)(m0 >> 11) * HID * SEQ;
    gemm_f32src<OUT_BF16_T>(v, HID, WV, HID, HID, outb, SEQ, 1.0f,
                            m0, n0, m0 & (SEQ - 1));
  }
}

// ---------- kernel 2: scores = (q @ k^T) / sqrt(H), bf16 (TDM pipeline) ----------
__global__ __launch_bounds__(256) void scores_kernel(
    const unsigned short* __restrict__ qb, const unsigned short* __restrict__ kb,
    unsigned short* __restrict__ sc)
{
  const int m0 = blockIdx.x * 128, n0 = blockIdx.y * 128, z = blockIdx.z;
  const unsigned short* A = qb + (size_t)z * SEQ * HID;
  const unsigned short* B = kb + (size_t)z * SEQ * HID;
  unsigned short*       C = sc + (size_t)z * SEQ * SEQ;
  gemm_bf16_tdm<OUT_BF16>(A, HID, B, HID, HID, C, SEQ, 0.03125f, m0, n0);
}

// ---------- kernel 3: row softmax, in place (fp32 math) ----------
__global__ __launch_bounds__(256) void softmax_kernel(unsigned short* __restrict__ sc)
{
  __shared__ float red[256];
  const int tid = threadIdx.x;
  unsigned short* row = sc + (size_t)blockIdx.x * SEQ;

  uint4 d = *(const uint4*)(row + tid * 8);
  unsigned dv[4] = {d.x, d.y, d.z, d.w};
  float x[8];
#pragma unroll
  for (int j = 0; j < 4; ++j) {
    x[2 * j]     = bf2f((unsigned short)(dv[j] & 0xFFFFu));
    x[2 * j + 1] = bf2f((unsigned short)(dv[j] >> 16));
  }
  float m = x[0];
#pragma unroll
  for (int j = 1; j < 8; ++j) m = fmaxf(m, x[j]);
  red[tid] = m; __syncthreads();
  for (int s = 128; s > 0; s >>= 1) {
    if (tid < s) red[tid] = fmaxf(red[tid], red[tid + s]);
    __syncthreads();
  }
  const float rmax = red[0];
  __syncthreads();

  float p[8], lsum = 0.0f;
#pragma unroll
  for (int j = 0; j < 8; ++j) { p[j] = __expf(x[j] - rmax); lsum += p[j]; }
  red[tid] = lsum; __syncthreads();
  for (int s = 128; s > 0; s >>= 1) {
    if (tid < s) red[tid] += red[tid + s];
    __syncthreads();
  }
  const float inv = 1.0f / red[0];

  unsigned ov[4];
#pragma unroll
  for (int j = 0; j < 4; ++j) ov[j] = f2bf_pk(p[2 * j] * inv, p[2 * j + 1] * inv);
  uint4 o; o.x = ov[0]; o.y = ov[1]; o.z = ov[2]; o.w = ov[3];
  *(uint4*)(row + tid * 8) = o;
}

// ---------- kernel 4: out = P @ V  (B operand = vt, NT layout, async pipeline) ----------
__global__ __launch_bounds__(256) void pv_kernel(
    const unsigned short* __restrict__ sc, const unsigned short* __restrict__ vt,
    float* __restrict__ out)
{
  const int m0 = blockIdx.x * 128, n0 = blockIdx.y * 128, z = blockIdx.z;
  const unsigned short* A = sc + (size_t)z * SEQ * SEQ;
  const unsigned short* B = vt + (size_t)z * HID * SEQ;
  float*                C = out + (size_t)z * SEQ * HID;
  gemm_bf16_async<OUT_F32>(A, SEQ, B, SEQ, SEQ, C, HID, 1.0f, m0, n0);
}

// ---------- host launcher ----------
extern "C" void kernel_launch(void* const* d_in, const int* in_sizes, int n_in,
                              void* d_out, int out_size, void* d_ws, size_t ws_size,
                              hipStream_t stream) {
  (void)in_sizes; (void)n_in; (void)out_size; (void)ws_size;
  const float* query = (const float*)d_in[0];
  const float* key_  = (const float*)d_in[1];
  const float* value = (const float*)d_in[2];
  const float* WQ    = (const float*)d_in[3];
  const float* WK    = (const float*)d_in[4];
  const float* WV    = (const float*)d_in[5];
  float* out = (float*)d_out;

  // ws layout (bf16): qb 16MB | kb 16MB | vt 16MB | scores 32MB  = 80MB
  unsigned short* qb = (unsigned short*)d_ws;
  unsigned short* kb = qb + (size_t)BATCH * SEQ * HID;
  unsigned short* vt = kb + (size_t)BATCH * SEQ * HID;
  unsigned short* sc = vt + (size_t)BATCH * SEQ * HID;

  dim3 blk(256);
  dim3 g1((BATCH * SEQ) / 128, HID / 128, 3);
  hipLaunchKernelGGL(qkv_kernel, g1, blk, 0, stream,
                     query, key_, value, WQ, WK, WV, qb, kb, vt);

  dim3 g2(SEQ / 128, SEQ / 128, BATCH);
  hipLaunchKernelGGL(scores_kernel, g2, blk, 0, stream, qb, kb, sc);

  hipLaunchKernelGGL(softmax_kernel, dim3(BATCH * SEQ), blk, 0, stream, sc);

  dim3 g4(SEQ / 128, HID / 128, BATCH);
  hipLaunchKernelGGL(pv_kernel, g4, blk, 0, stream, sc, vt, out);
}
